// lstmModel_70987219468378
// MI455X (gfx1250) — matrix operand order repaired and checked by reading, not compiled, not run on hardware
//
#include <hip/hip_runtime.h>
#include <hip/hip_bf16.h>
#include <math.h>

// ---------------------------------------------------------------------------
// Problem constants (from the reference)
// ---------------------------------------------------------------------------
#define BATCH   64
#define F_IN    28
#define T_LEN   8192
#define C1_CH   8
#define NF_CH   16
#define KSIZE   9
#define STRIDE  3
#define TP_LEN  2728          // (8192 - 9)/3 + 1
#define W_MAX   128
#define C_LSTM  128           // NF*C1
#define H_DIM   32
#define G_DIM   128           // 4*H
#define EPS_BN  1e-5f

typedef __attribute__((ext_vector_type(16))) _Float16 v16h;
typedef __attribute__((ext_vector_type(8)))  _Float16 v8h;
typedef __attribute__((ext_vector_type(8)))  float    v8f;

// ---------------------------------------------------------------------------
// Kernel 0: preparation.
//  - Fold BN into (scale, shift).
//  - bias[j] = b_ih[j] + b_hh[j].
//  - Pack w_ih^T directly into WMMA B-fragment order:
//      wBfrag[((tileN*4 + kk)*32 + lane)*16 + h] = w_ih[n][k]   (as f16)
//    where n = tileN*16 + (lane & 15),
//          k = kk*32 + (lane < 16 ? 0 : 16) + h      (contiguous K order, same as A).
//    The GEMM then loads each fragment as one aligned 32-byte vector.
// ---------------------------------------------------------------------------
__global__ void prep_kernel(const float* __restrict__ w_ih,
                            const float* __restrict__ b_ih,
                            const float* __restrict__ b_hh,
                            const float* __restrict__ bn_gamma,
                            const float* __restrict__ bn_beta,
                            const float* __restrict__ bn_mean,
                            const float* __restrict__ bn_var,
                            _Float16*    __restrict__ wBfrag,  // [16384] f16
                            float*       __restrict__ bias,    // [128]
                            float*       __restrict__ scsh)    // [32] scale|shift
{
    int tid = blockIdx.x * 256 + threadIdx.x;
    if (tid < 8 * 4 * 32 * 16) {
        const int h     = tid & 15;
        const int lane  = (tid >> 4) & 31;
        const int kk    = (tid >> 9) & 3;
        const int tileN = tid >> 11;
        const int n = tileN * 16 + (lane & 15);
        const int k = kk * 32 + ((lane < 16) ? 0 : 16) + h;
        wBfrag[tid] = (_Float16)w_ih[n * C_LSTM + k];
    }
    if (tid < G_DIM) bias[tid] = b_ih[tid] + b_hh[tid];
    if (tid < NF_CH) {
        float sc = bn_gamma[tid] * rsqrtf(bn_var[tid] + EPS_BN);
        scsh[tid]         = sc;
        scsh[NF_CH + tid] = bn_beta[tid] - bn_mean[tid] * sc;
    }
}

// ---------------------------------------------------------------------------
// Kernel 1: pointwise conv (28 -> 8) + ReLU.  y1[b][c1][t], coalesced over t.
// ---------------------------------------------------------------------------
__global__ void conv1_kernel(const float* __restrict__ x,
                             const float* __restrict__ w1,   // [8][28]
                             const float* __restrict__ b1,   // [8]
                             float*       __restrict__ y1)   // [B][8][T]
{
    __shared__ float w1s[C1_CH * F_IN];
    __shared__ float b1s[C1_CH];
    if (threadIdx.x < C1_CH * F_IN) w1s[threadIdx.x] = w1[threadIdx.x];
    if (threadIdx.x < C1_CH)        b1s[threadIdx.x] = b1[threadIdx.x];
    __syncthreads();

    const size_t gid = (size_t)blockIdx.x * 256 + threadIdx.x;  // B*T threads
    const int b = (int)(gid >> 13);   // / 8192
    const int t = (int)(gid & (T_LEN - 1));

    float xin[F_IN];
    const float* xb = x + ((size_t)b * F_IN) * T_LEN + t;
#pragma unroll
    for (int c = 0; c < F_IN; ++c) xin[c] = xb[(size_t)c * T_LEN];

#pragma unroll
    for (int o = 0; o < C1_CH; ++o) {
        float v = b1s[o];
#pragma unroll
        for (int c = 0; c < F_IN; ++c) v = fmaf(w1s[o * F_IN + c], xin[c], v);
        y1[((size_t)b * C1_CH + o) * T_LEN + t] = fmaxf(v, 0.0f);
    }
}

// ---------------------------------------------------------------------------
// Kernel 2: per-word segment mean pool, recomputing conv2+BN+ReLU on the fly
// only for timesteps inside [onset, offset).  One block per (b, w), one
// thread per LSTM channel c = nf*8 + c1 (torch reshape order).
// Emits pooled in f16 for the WMMA GEMM: pooled[(b*W+w)*128 + c].
// ---------------------------------------------------------------------------
__global__ void pool_kernel(const float* __restrict__ y1,     // [B][8][T]
                            const float* __restrict__ w2,     // [16][9]
                            const float* __restrict__ b2,     // [16]
                            const float* __restrict__ scsh,   // [32]
                            const int*   __restrict__ onset,  // [B][W]
                            const int*   __restrict__ offset, // [B][W]
                            _Float16*    __restrict__ pooled) // [B*W][128] f16
{
    __shared__ float w2s[NF_CH * KSIZE];   // 144
    __shared__ float scs[NF_CH], shs[NF_CH], b2s[NF_CH];
    const int tid = threadIdx.x;           // 0..127
    if (tid < 128)                  w2s[tid]       = w2[tid];
    if (tid < NF_CH * KSIZE - 128)  w2s[128 + tid] = w2[128 + tid];
    if (tid < NF_CH) {
        scs[tid] = scsh[tid];
        shs[tid] = scsh[NF_CH + tid];
        b2s[tid] = b2[tid];
    }
    __syncthreads();

    const int b  = blockIdx.y;
    const int w  = blockIdx.x;
    const int c  = tid;
    const int nf = c >> 3;
    const int c1 = c & 7;

    const int on  = onset [b * W_MAX + w];
    const int off = offset[b * W_MAX + w];

    const float* yb = y1 + ((size_t)b * C1_CH + c1) * T_LEN;
    float sum = 0.0f;
    for (int t = on; t < off; ++t) {
        float v = b2s[nf];
        const int t0 = STRIDE * t;
#pragma unroll
        for (int k = 0; k < KSIZE; ++k) v = fmaf(w2s[nf * KSIZE + k], yb[t0 + k], v);
        v = fmaf(v, scs[nf], shs[nf]);     // BN eval
        sum += fmaxf(v, 0.0f);             // ReLU then accumulate
    }
    const float len = (float)max(off - on, 1);
    pooled[((size_t)b * W_MAX + w) * C_LSTM + c] = (_Float16)(sum / len);
}

// ---------------------------------------------------------------------------
// Kernel 3: WMMA GEMM  gatesX[m][n] = sum_k pooled[m][k] * w_ih[n][k]
// M = B*W = 8192, N = 128, K = 128.  v_wmma_f32_16x16x32_f16, f32 accumulate.
// 64 blocks x 8 waves; each wave owns one 16-row M tile and the FULL N=128
// (8 accumulators), reusing one A fragment across all 8 N-tiles.
// B is pre-packed into fragment order and staged in LDS (32 KB) once per
// block; fragments come back as single ds_load_b128 pairs.
// Static mix per wave: 4 kk * (2 global b128 A loads + 8 * (LDS b128 frag +
// v_wmma)) -> 32 WMMA instructions.
// ---------------------------------------------------------------------------
__global__ void gemm_gates_kernel(const _Float16* __restrict__ A,       // [8192][128]
                                  const _Float16* __restrict__ wBfrag,  // [16384]
                                  float*          __restrict__ D)       // [8192][128]
{
    __shared__ _Float16 Bs[8 * 4 * 32 * 16];   // 32 KB, fragment order

    // Cooperative staging of packed B: 16384 halves / 256 threads = 8 v8h each
    {
        const v8h* src = (const v8h*)wBfrag;
        v8h*       dst = (v8h*)Bs;
#pragma unroll
        for (int i = 0; i < 8; ++i) {
            dst[threadIdx.x + 256 * i] = src[threadIdx.x + 256 * i];
        }
    }
    __syncthreads();

    const int lane  = threadIdx.x & 31;
    const int wave  = threadIdx.x >> 5;
    const int tileM = blockIdx.x * 8 + wave;   // 512 M tiles

    const int rowA = lane & 15;
    const int kbA  = (lane < 16) ? 0 : 16;     // contiguous K layout (matches packed B)

    v8f acc[8];
#pragma unroll
    for (int tn = 0; tn < 8; ++tn) acc[tn] = (v8f){};

    const _Float16* Arow = A + ((size_t)tileM * 16 + rowA) * C_LSTM;

#pragma unroll
    for (int kk = 0; kk < 4; ++kk) {
        const int k0 = kk * 32;

        // A fragment: lanes<16 hold K[0..15]; lanes>=16 K[16..31]
        const _Float16* Ap = Arow + k0 + kbA;
        v8h alo = *(const v8h*)(Ap);
        v8h ahi = *(const v8h*)(Ap + 8);
        v16h a;
#pragma unroll
        for (int h = 0; h < 8; ++h) { a[h] = alo[h]; a[h + 8] = ahi[h]; }

#pragma unroll
        for (int tn = 0; tn < 8; ++tn) {
            // B fragment: one aligned 32-byte LDS read per (tn, kk)
            const v16h bfrag =
                *(const v16h*)(Bs + (((tn * 4 + kk) * 32) + lane) * 16);
            acc[tn] = __builtin_amdgcn_wmma_f32_16x16x32_f16(
                          false, a, false, bfrag, (short)0, acc[tn], false, false);
        }
    }

    // D layout: VGPR r -> M = 16*tileM + r + (lane<16 ? 0 : 8), N = 16*tn + (lane&15)
    const int mBase = tileM * 16 + ((lane < 16) ? 0 : 8);
    const int nOff  = (lane & 15);
#pragma unroll
    for (int tn = 0; tn < 8; ++tn) {
        const int n = tn * 16 + nOff;
#pragma unroll
        for (int r = 0; r < 8; ++r) {
            D[(size_t)(mBase + r) * G_DIM + n] = acc[tn][r];
        }
    }
}

// ---------------------------------------------------------------------------
// Kernel 4: masked LSTM recurrence.  One block per batch element, 128 threads
// (one per gate).  w_hh row lives in registers; h in LDS; c in registers of
// threads 0..31.  gatesX already holds x_t @ w_ih^T.
// ---------------------------------------------------------------------------
__global__ void lstm_kernel(const float* __restrict__ gatesX, // [B][W][128]
                            const float* __restrict__ w_hh,   // [128][32]
                            const float* __restrict__ bias,   // [128]
                            const int*   __restrict__ s,      // [B]
                            float*       __restrict__ out)    // [B][32]
{
    __shared__ float hS[H_DIM];
    __shared__ float gS[G_DIM];

    const int b = blockIdx.x;
    const int j = threadIdx.x;            // 0..127

    float wrow[H_DIM];
#pragma unroll
    for (int k = 0; k < H_DIM; ++k) wrow[k] = w_hh[j * H_DIM + k];
    const float bj = bias[j];
    const int  sb = s[b];

    if (j < H_DIM) hS[j] = 0.0f;
    float cstate = 0.0f;
    __syncthreads();

    const float* gx = gatesX + (size_t)b * W_MAX * G_DIM;
    for (int w = 0; w < W_MAX; ++w) {
        float g = gx[(size_t)w * G_DIM + j] + bj;
#pragma unroll
        for (int k = 0; k < H_DIM; ++k) g = fmaf(hS[k], wrow[k], g);
        gS[j] = g;
        __syncthreads();
        if (j < H_DIM) {
            float ig = 1.0f / (1.0f + __expf(-gS[j]));
            float fg = 1.0f / (1.0f + __expf(-gS[H_DIM + j]));
            float gg = tanhf(gS[2 * H_DIM + j]);
            float og = 1.0f / (1.0f + __expf(-gS[3 * H_DIM + j]));
            float cn = fmaf(fg, cstate, ig * gg);
            float hn = og * tanhf(cn);
            if (w < sb) { cstate = cn; hS[j] = hn; }
        }
        __syncthreads();
    }
    if (j < H_DIM) out[b * H_DIM + j] = hS[j];
}

// ---------------------------------------------------------------------------
// Launcher
// ---------------------------------------------------------------------------
extern "C" void kernel_launch(void* const* d_in, const int* in_sizes, int n_in,
                              void* d_out, int out_size, void* d_ws, size_t ws_size,
                              hipStream_t stream) {
    const float* x        = (const float*)d_in[0];
    const int*   s        = (const int*)  d_in[1];
    const int*   onset    = (const int*)  d_in[2];
    const int*   offset   = (const int*)  d_in[3];
    const float* w1       = (const float*)d_in[4];
    const float* b1       = (const float*)d_in[5];
    const float* w2       = (const float*)d_in[6];
    const float* b2       = (const float*)d_in[7];
    const float* bn_gamma = (const float*)d_in[8];
    const float* bn_beta  = (const float*)d_in[9];
    const float* bn_mean  = (const float*)d_in[10];
    const float* bn_var   = (const float*)d_in[11];
    const float* w_ih     = (const float*)d_in[12];
    const float* w_hh     = (const float*)d_in[13];
    const float* b_ih     = (const float*)d_in[14];
    const float* b_hh     = (const float*)d_in[15];
    float* out = (float*)d_out;

    // Workspace layout (all offsets 256B-aligned)
    char* ws = (char*)d_ws;
    float*    y1     = (float*)   (ws + 0);                    // 64*8*8192*4   = 16,777,216
    _Float16* pooled = (_Float16*)(ws + 16777216);             // 8192*128*2    =  2,097,152
    _Float16* wBfrag = (_Float16*)(ws + 18874368);             // 16384*2       =     32,768
    float*    bias   = (float*)   (ws + 18907136);             // 128*4         =        512
    float*    scsh   = (float*)   (ws + 18907648);             // 32*4 -> pad   =        256
    float*    gatesX = (float*)   (ws + 18907904);             // 8192*128*4    =  4,194,304
    // total: 23,102,208 bytes

    // 0) prep: BN fold, w_ih^T -> f16 fragment pack, bias sum
    prep_kernel<<<64, 256, 0, stream>>>(w_ih, b_ih, b_hh,
                                        bn_gamma, bn_beta, bn_mean, bn_var,
                                        wBfrag, bias, scsh);

    // 1) pointwise conv + ReLU : B*T threads
    conv1_kernel<<<(BATCH * T_LEN) / 256, 256, 0, stream>>>(x, w1, b1, y1);

    // 2) segment pooling with on-the-fly conv2+BN+ReLU : block per (w, b)
    pool_kernel<<<dim3(W_MAX, BATCH), C_LSTM, 0, stream>>>(y1, w2, b2, scsh,
                                                           onset, offset, pooled);

    // 3) WMMA GEMM: gatesX = pooled @ w_ih^T  (8192 x 128 x 128)
    //    64 blocks x 8 waves, one 16x128 output strip per wave.
    gemm_gates_kernel<<<64, 256, 0, stream>>>(pooled, wBfrag, gatesX);

    // 4) masked LSTM recurrence -> final hidden state [64, 32]
    lstm_kernel<<<BATCH, G_DIM, 0, stream>>>(gatesX, w_hh, bias, s, out);
}
